// Level3AffectivePatterns_40295383171427
// MI455X (gfx1250) — matrix operand order repaired and checked by reading, not compiled, not run on hardware
//
#include <hip/hip_runtime.h>
#include <hip/hip_bf16.h>
#include <math.h>

// ---------------- types for WMMA ----------------
typedef __attribute__((ext_vector_type(16))) __bf16 v16bf;
typedef __attribute__((ext_vector_type(8)))  float  v8f;
typedef __attribute__((ext_vector_type(4)))  int    i4;

#define WG_GEMM 256
#define BM 128
#define BN 128
#define BK 32
#define LDT 40   // LDS row stride in bf16 elems (80 bytes, multiple of 16B)

#if __has_builtin(__builtin_amdgcn_global_load_async_to_lds_b128) && \
    __has_builtin(__builtin_amdgcn_s_wait_asynccnt)
#define HAVE_ASYNC 1
#else
#define HAVE_ASYNC 0
#endif

__device__ __forceinline__ int imin(int a, int b) { return a < b ? a : b; }

__device__ __forceinline__ unsigned short f2bf(float f) {
    unsigned int u = __float_as_uint(f);
    unsigned int r = u + 0x7FFFu + ((u >> 16) & 1u);
    return (unsigned short)(r >> 16);
}
__device__ __forceinline__ float sigmoidf_(float x) { return 1.0f / (1.0f + expf(-x)); }
__device__ __forceinline__ float geluf_(float x) { return 0.5f * x * (1.0f + erff(x * 0.70710678118654752f)); }

#if HAVE_ASYNC
// 16-byte async copy: global -> LDS, tracked by ASYNCcnt (no VGPR round trip)
__device__ __forceinline__ void cp_async16(void* lds, const void* g) {
    __builtin_amdgcn_global_load_async_to_lds_b128(
        (__attribute__((address_space(1))) i4*)(i4*)(void*)g,
        (__attribute__((address_space(3))) i4*)(i4*)lds, 0, 0);
}
#endif

// ---------------- generic WMMA GEMM: C[M,N] = A[M,K] (bf16) * B[N,K]^T (bf16) + bias, act ----------------
// act: 0 = none, 1 = gelu(exact), 2 = sigmoid
__global__ __launch_bounds__(WG_GEMM)
void k_gemm(const unsigned short* __restrict__ A, int lda,
            const unsigned short* __restrict__ Bw, int ldb,
            const float* __restrict__ bias,
            float* __restrict__ C, int ldc,
            int M, int N, int K, int act)
{
    __shared__ alignas(16) unsigned short sA[2][BM * LDT];
    __shared__ alignas(16) unsigned short sB[2][BN * LDT];

    const int tid  = threadIdx.x;
    const int lane = tid & 31;
    const int wave = tid >> 5;           // 0..7
    const int wm   = wave >> 2;          // 0..1  (64 rows each)
    const int wn   = wave & 3;           // 0..3  (32 cols each)
    const int l16  = lane & 15;
    const int lhi  = lane >> 4;          // 0/1
    const int bm0  = blockIdx.y * BM;
    const int bn0  = blockIdx.x * BN;

    const int row_c = tid >> 1;          // 0..127 (tile row staged by this thread)
    const int seg   = tid & 1;           // 0/1 (16 bf16 each)

    v8f acc[4][2];
#pragma unroll
    for (int mt = 0; mt < 4; ++mt)
#pragma unroll
        for (int nt = 0; nt < 2; ++nt)
            acc[mt][nt] = (v8f){0.f,0.f,0.f,0.f,0.f,0.f,0.f,0.f};

    const int ga = imin(bm0 + row_c, M - 1);
    const int gb = imin(bn0 + row_c, N - 1);
    const unsigned short* gArow = A  + (size_t)ga * lda + seg * 16;
    const unsigned short* gBrow = Bw + (size_t)gb * ldb + seg * 16;
    unsigned short* lA = &sA[0][0] + row_c * LDT + seg * 16;
    unsigned short* lB = &sB[0][0] + row_c * LDT + seg * 16;
    const int bufstrideA = BM * LDT;
    const int bufstrideB = BN * LDT;

    const int nk = K / BK;

#if HAVE_ASYNC
    // -------- double-buffered async pipeline --------
    // prologue: stage tile 0 into buffer 0 (4 x b128 per thread)
    {
        cp_async16(lA,      gArow);
        cp_async16(lA + 8,  gArow + 8);
        cp_async16(lB,      gBrow);
        cp_async16(lB + 8,  gBrow + 8);
    }
    for (int i = 0; i < nk; ++i) {
        const int cur = i & 1;
        if (i + 1 < nk) {
            const int nxt = (i + 1) & 1;
            const unsigned short* gan = gArow + (size_t)(i + 1) * BK;
            const unsigned short* gbn = gBrow + (size_t)(i + 1) * BK;
            cp_async16(lA + nxt * bufstrideA,     gan);
            cp_async16(lA + nxt * bufstrideA + 8, gan + 8);
            cp_async16(lB + nxt * bufstrideB,     gbn);
            cp_async16(lB + nxt * bufstrideB + 8, gbn + 8);
            __builtin_amdgcn_s_wait_asynccnt(4);   // tile i's group complete (in-order)
        } else {
            __builtin_amdgcn_s_wait_asynccnt(0);
        }
        __syncthreads();

        const unsigned short* cA = &sA[0][0] + cur * bufstrideA;
        const unsigned short* cB = &sB[0][0] + cur * bufstrideB;
        v16bf bfrag[2];
#pragma unroll
        for (int nt = 0; nt < 2; ++nt) {
            const uint4* p = (const uint4*)(cB + (wn * 32 + nt * 16 + l16) * LDT + lhi * 16);
            ((uint4*)&bfrag[nt])[0] = p[0];
            ((uint4*)&bfrag[nt])[1] = p[1];
        }
#pragma unroll
        for (int mt = 0; mt < 4; ++mt) {
            v16bf afrag;
            const unsigned short* ar = cA + (wm * 64 + mt * 16 + l16) * LDT;
            ((uint4*)&afrag)[0] = *(const uint4*)(ar + lhi * 16);
            ((uint4*)&afrag)[1] = *(const uint4*)(ar + lhi * 16 + 8);
#pragma unroll
            for (int nt = 0; nt < 2; ++nt)
                acc[mt][nt] = __builtin_amdgcn_wmma_f32_16x16x32_bf16(
                    false, afrag, false, bfrag[nt], (short)0, acc[mt][nt], false, false);
        }
        __syncthreads();   // all waves done reading buffer `cur` before it is refilled
    }
#else
    // -------- fallback: synchronous staging --------
    for (int i = 0; i < nk; ++i) {
        const int k0 = i * BK;
        {
            const uint4* srcA = (const uint4*)(gArow + k0);
            uint4* dstA = (uint4*)lA;
            dstA[0] = srcA[0]; dstA[1] = srcA[1];
            const uint4* srcB = (const uint4*)(gBrow + k0);
            uint4* dstB = (uint4*)lB;
            dstB[0] = srcB[0]; dstB[1] = srcB[1];
        }
        if (k0 + BK < K) {
            __builtin_prefetch(gArow + k0 + BK, 0, 1);
            __builtin_prefetch(gBrow + k0 + BK, 0, 1);
        }
        __syncthreads();
        v16bf bfrag[2];
#pragma unroll
        for (int nt = 0; nt < 2; ++nt) {
            const uint4* p = (const uint4*)(&sB[0][0] + (wn * 32 + nt * 16 + l16) * LDT + lhi * 16);
            ((uint4*)&bfrag[nt])[0] = p[0];
            ((uint4*)&bfrag[nt])[1] = p[1];
        }
#pragma unroll
        for (int mt = 0; mt < 4; ++mt) {
            v16bf afrag;
            const unsigned short* ar = &sA[0][0] + (wm * 64 + mt * 16 + l16) * LDT;
            ((uint4*)&afrag)[0] = *(const uint4*)(ar + lhi * 16);
            ((uint4*)&afrag)[1] = *(const uint4*)(ar + lhi * 16 + 8);
#pragma unroll
            for (int nt = 0; nt < 2; ++nt)
                acc[mt][nt] = __builtin_amdgcn_wmma_f32_16x16x32_bf16(
                    false, afrag, false, bfrag[nt], (short)0, acc[mt][nt], false, false);
        }
        __syncthreads();
    }
#endif

    // ---- epilogue: C/D layout: lane -> n=l16, m=r+8*lhi
#pragma unroll
    for (int mt = 0; mt < 4; ++mt) {
#pragma unroll
        for (int nt = 0; nt < 2; ++nt) {
            const int n = bn0 + wn * 32 + nt * 16 + l16;
            if (n >= N) continue;
            const float bv = bias ? bias[n] : 0.0f;
#pragma unroll
            for (int r = 0; r < 8; ++r) {
                const int m = bm0 + wm * 64 + mt * 16 + r + 8 * lhi;
                if (m >= M) continue;
                float v = acc[mt][nt][r] + bv;
                if (act == 1)      v = geluf_(v);
                else if (act == 2) v = sigmoidf_(v);
                C[(size_t)m * ldc + n] = v;
            }
        }
    }
}

// ---------------- LSTM step: g = xg[:,t,:] + h_prev @ whh^T ; gate math ; h -> hseq[:,t,:] ----------------
// grid: 64 blocks (16 h-units each), 128 threads = 4 waves (wave g = gate g)
__global__ __launch_bounds__(128)
void k_lstm_step(const unsigned short* __restrict__ hprev, int hp_stride,   // row b at hprev + b*hp_stride
                 const unsigned short* __restrict__ whh,                    // [4H][H] bf16
                 const float* __restrict__ xg,                              // [B*T][4H]
                 float* __restrict__ c,                                     // [16][H]
                 unsigned short* __restrict__ hseq,                         // row (b*T+t), [..][H]
                 int t)
{
    __shared__ float sg[4 * 16 * 16];
    const int tid  = threadIdx.x;
    const int lane = tid & 31;
    const int g    = tid >> 5;           // gate 0..3 (i,f,g,o)
    const int l16  = lane & 15;
    const int lhi  = lane >> 4;
    const int u0   = blockIdx.x * 16;
    const int n0   = g * 1024 + u0;

    v8f acc = (v8f){0.f,0.f,0.f,0.f,0.f,0.f,0.f,0.f};

    const unsigned short* hrow = hprev + (size_t)l16 * hp_stride;  // A row m = batch = l16
    const unsigned short* wrow = whh + (size_t)(n0 + l16) * 1024;  // B row n

    for (int k0 = 0; k0 < 1024; k0 += 32) {
        v16bf afrag, bfrag;
        ((uint4*)&afrag)[0] = *(const uint4*)(hrow + k0 + lhi * 16);
        ((uint4*)&afrag)[1] = *(const uint4*)(hrow + k0 + lhi * 16 + 8);
        ((uint4*)&bfrag)[0] = *(const uint4*)(wrow + k0 + lhi * 16);
        ((uint4*)&bfrag)[1] = *(const uint4*)(wrow + k0 + lhi * 16 + 8);
        acc = __builtin_amdgcn_wmma_f32_16x16x32_bf16(false, afrag, false, bfrag,
                                                      (short)0, acc, false, false);
    }

    // add xg and park gate pre-activations in LDS
#pragma unroll
    for (int r = 0; r < 8; ++r) {
        const int b = r + 8 * lhi;                       // batch
        const int n = n0 + l16;                          // gate column
        const float v = acc[r] + xg[((size_t)b * 512 + t) * 4096 + n];
        sg[g * 256 + b * 16 + l16] = v;
    }
    __syncthreads();

    // 256 cells / 128 threads
#pragma unroll
    for (int p = 0; p < 2; ++p) {
        const int idx = tid + p * 128;
        const int b = idx >> 4;
        const int u = idx & 15;
        const float iv = sg[0 * 256 + b * 16 + u];
        const float fv = sg[1 * 256 + b * 16 + u];
        const float gv = sg[2 * 256 + b * 16 + u];
        const float ov = sg[3 * 256 + b * 16 + u];
        float cc = c[(size_t)b * 1024 + u0 + u];
        cc = sigmoidf_(fv) * cc + sigmoidf_(iv) * tanhf(gv);
        c[(size_t)b * 1024 + u0 + u] = cc;
        const float h = sigmoidf_(ov) * tanhf(cc);
        hseq[((size_t)b * 512 + t) * 1024 + u0 + u] = f2bf(h);
    }
}

// ---------------- LayerNorm + GELU (row-wise) -> bf16 ----------------
__global__ __launch_bounds__(256)
void k_ln_gelu(const float* __restrict__ x, const float* __restrict__ g,
               const float* __restrict__ b, unsigned short* __restrict__ out, int D)
{
    __shared__ float red[256];
    const int row = blockIdx.x;
    const int tid = threadIdx.x;
    const float* xr = x + (size_t)row * D;

    float s = 0.f;
    for (int j = tid; j < D; j += 256) s += xr[j];
    red[tid] = s; __syncthreads();
    for (int st = 128; st > 0; st >>= 1) { if (tid < st) red[tid] += red[tid + st]; __syncthreads(); }
    const float mu = red[0] / (float)D;
    __syncthreads();

    float v = 0.f;
    for (int j = tid; j < D; j += 256) { const float d = xr[j] - mu; v += d * d; }
    red[tid] = v; __syncthreads();
    for (int st = 128; st > 0; st >>= 1) { if (tid < st) red[tid] += red[tid + st]; __syncthreads(); }
    const float inv = rsqrtf(red[0] / (float)D + 1e-5f);

    for (int j = tid; j < D; j += 256) {
        float y = (xr[j] - mu) * inv * g[j] + b[j];
        out[(size_t)row * D + j] = f2bf(geluf_(y));
    }
}

// ---------------- attention softmax over M=100, write zero-padded bf16 [.,128] ----------------
__global__ __launch_bounds__(128)
void k_softmax(const float* __restrict__ s, unsigned short* __restrict__ attn)
{
    __shared__ float red[128];
    const int row = blockIdx.x;
    const int tid = threadIdx.x;
    const float scale = 0.04419417382415922f; // 1/sqrt(512)
    float v = (tid < 100) ? s[(size_t)row * 128 + tid] * scale : -1e30f;
    red[tid] = v; __syncthreads();
    for (int st = 64; st > 0; st >>= 1) { if (tid < st) red[tid] = fmaxf(red[tid], red[tid + st]); __syncthreads(); }
    const float mx = red[0]; __syncthreads();
    float e = (tid < 100) ? expf(v - mx) : 0.f;
    red[tid] = e; __syncthreads();
    for (int st = 64; st > 0; st >>= 1) { if (tid < st) red[tid] += red[tid + st]; __syncthreads(); }
    attn[(size_t)row * 128 + tid] = f2bf(e / red[0]);
}

// ---------------- mood (3) + pers (5) heads ----------------
__global__ __launch_bounds__(128)
void k_heads(const float* __restrict__ enc,
             const float* __restrict__ mw, const float* __restrict__ mb,
             const float* __restrict__ pw, const float* __restrict__ pb,
             float* __restrict__ mood, float* __restrict__ pers)
{
    __shared__ float red[8][128];
    const int row = blockIdx.x;
    const int tid = threadIdx.x;
    const float* er = enc + (size_t)row * 1024;
    float acc[8] = {0.f,0.f,0.f,0.f,0.f,0.f,0.f,0.f};
    for (int k = tid; k < 1024; k += 128) {
        const float e = er[k];
        acc[0] += e * mw[k];
        acc[1] += e * mw[1024 + k];
        acc[2] += e * mw[2048 + k];
        acc[3] += e * pw[k];
        acc[4] += e * pw[1024 + k];
        acc[5] += e * pw[2048 + k];
        acc[6] += e * pw[3072 + k];
        acc[7] += e * pw[4096 + k];
    }
#pragma unroll
    for (int j = 0; j < 8; ++j) red[j][tid] = acc[j];
    __syncthreads();
    for (int st = 64; st > 0; st >>= 1) {
        if (tid < st) {
#pragma unroll
            for (int j = 0; j < 8; ++j) red[j][tid] += red[j][tid + st];
        }
        __syncthreads();
    }
    if (tid < 3)      mood[(size_t)row * 3 + tid] = red[tid][0] + mb[tid];
    else if (tid < 8) pers[(size_t)row * 5 + (tid - 3)] = sigmoidf_(red[tid][0] + pb[tid - 3]);
}

// ---------------- small utility kernels ----------------
__global__ void k_cvt(const float* __restrict__ in, unsigned short* __restrict__ out, size_t n) {
    size_t i = (size_t)blockIdx.x * 256 + threadIdx.x;
    if (i < n) out[i] = f2bf(in[i]);
}
__global__ void k_zero32(unsigned int* __restrict__ p, size_t n) {
    size_t i = (size_t)blockIdx.x * 256 + threadIdx.x;
    if (i < n) p[i] = 0u;
}
__global__ void k_addvec(const float* __restrict__ a, const float* __restrict__ b,
                         float* __restrict__ o, int n) {
    int i = blockIdx.x * 256 + threadIdx.x;
    if (i < n) o[i] = a[i] + b[i];
}
__global__ void k_concat(const float* __restrict__ z3, const float* __restrict__ rd,
                         unsigned short* __restrict__ x0, size_t n) {
    size_t i = (size_t)blockIdx.x * 256 + threadIdx.x;
    if (i >= n) return;
    size_t r = i / 1536, j = i % 1536;
    float v = (j < 1024) ? z3[r * 1024 + j] : rd[r * 512 + (j - 1024)];
    x0[i] = f2bf(v);
}
// values (100x512) -> valT_bf (512x128, zero pad)
__global__ void k_valT(const float* __restrict__ vals, unsigned short* __restrict__ vT) {
    int i = blockIdx.x * 256 + threadIdx.x;   // over 512*128
    if (i >= 512 * 128) return;
    int d = i >> 7, m = i & 127;
    vT[i] = (m < 100) ? f2bf(vals[(size_t)m * 512 + d]) : (unsigned short)0;
}
// zf gather: enc[(b*512+511), :] -> zf_bf (128x1024, zero pad rows)
__global__ void k_zf(const float* __restrict__ enc, unsigned short* __restrict__ zf) {
    int i = blockIdx.x * 256 + threadIdx.x;   // over 128*1024
    if (i >= 128 * 1024) return;
    int r = i >> 10, j = i & 1023;
    zf[i] = (r < 16) ? f2bf(enc[((size_t)r * 512 + 511) * 1024 + j]) : (unsigned short)0;
}
// t1 f32 (16x2048) -> t1_bf (128x2048, zero pad rows)
__global__ void k_t1pad(const float* __restrict__ t1, unsigned short* __restrict__ o) {
    int i = blockIdx.x * 256 + threadIdx.x;   // over 128*2048
    if (i >= 128 * 2048) return;
    int r = i >> 11, j = i & 2047;
    o[i] = (r < 16) ? f2bf(t1[(size_t)r * 2048 + j]) : (unsigned short)0;
}

// ---------------- host orchestration ----------------
static inline void launch_cvt(const float* in, unsigned short* out, size_t n, hipStream_t s) {
    k_cvt<<<dim3((unsigned)((n + 255) / 256)), dim3(256), 0, s>>>(in, out, n);
}
static inline void launch_gemm(const unsigned short* A, int lda, const unsigned short* B, int ldb,
                               const float* bias, float* C, int ldc,
                               int M, int N, int K, int act, hipStream_t s) {
    dim3 grid((N + BN - 1) / BN, (M + BM - 1) / BM);
    k_gemm<<<grid, dim3(WG_GEMM), 0, s>>>(A, lda, B, ldb, bias, C, ldc, M, N, K, act);
}

extern "C" void kernel_launch(void* const* d_in, const int* in_sizes, int n_in,
                              void* d_out, int out_size, void* d_ws, size_t ws_size,
                              hipStream_t stream) {
    (void)in_sizes; (void)n_in; (void)out_size; (void)ws_size;
    const int BT = 16 * 512;          // 8192 rows

    const float* z2     = (const float*)d_in[0];
    const float* memory = (const float*)d_in[1];
    const float* in_w   = (const float*)d_in[2];
    const float* in_b   = (const float*)d_in[3];
    const float* mq_w   = (const float*)d_in[4];
    const float* mq_b   = (const float*)d_in[5];
    const float* mk_w   = (const float*)d_in[6];
    const float* mk_b   = (const float*)d_in[7];
    const float* mv_w   = (const float*)d_in[8];
    const float* mv_b   = (const float*)d_in[9];
    const float* wih0   = (const float*)d_in[10];
    const float* bih0   = (const float*)d_in[11];
    const float* whh0   = (const float*)d_in[12];
    const float* bhh0   = (const float*)d_in[13];
    const float* wih1   = (const float*)d_in[14];
    const float* bih1   = (const float*)d_in[15];
    const float* whh1   = (const float*)d_in[16];
    const float* bhh1   = (const float*)d_in[17];
    const float* e1_w   = (const float*)d_in[18];
    const float* e1_b   = (const float*)d_in[19];
    const float* ln1_g  = (const float*)d_in[20];
    const float* ln1_b  = (const float*)d_in[21];
    const float* e2_w   = (const float*)d_in[22];
    const float* e2_b   = (const float*)d_in[23];
    const float* ln2_g  = (const float*)d_in[24];
    const float* ln2_b  = (const float*)d_in[25];
    const float* e3_w   = (const float*)d_in[26];
    const float* e3_b   = (const float*)d_in[27];
    const float* p1_w   = (const float*)d_in[28];
    const float* p1_b   = (const float*)d_in[29];
    const float* p2_w   = (const float*)d_in[30];
    const float* p2_b   = (const float*)d_in[31];
    const float* mood_w = (const float*)d_in[32];
    const float* mood_b = (const float*)d_in[33];
    const float* pers_w = (const float*)d_in[34];
    const float* pers_b = (const float*)d_in[35];

    float* out = (float*)d_out;
    float* out_enc  = out;                              // 8192*1024
    float* out_mood = out + (size_t)8192 * 1024;        // 8192*3
    float* out_pers = out_mood + (size_t)8192 * 3;      // 8192*5
    float* out_pred = out_pers + (size_t)8192 * 5;      // 16*1024

    // ---- workspace carve-up
    char* ws = (char*)d_ws;
    size_t off = 0;
    auto alloc = [&](size_t bytes) -> void* {
        void* p = ws + off;
        off += (bytes + 255) & ~(size_t)255;
        return p;
    };
    auto abf = [&](size_t elems) -> unsigned short* { return (unsigned short*)alloc(elems * 2); };
    auto af32 = [&](size_t elems) -> float* { return (float*)alloc(elems * 4); };

    // bf16 weight copies
    unsigned short* w_in  = abf((size_t)1024 * 512);
    unsigned short* w_mq  = abf((size_t)512 * 1024);
    unsigned short* w_mk  = abf((size_t)512 * 512);
    unsigned short* w_mv  = abf((size_t)512 * 512);
    unsigned short* w_ih0 = abf((size_t)4096 * 1536);
    unsigned short* w_hh0 = abf((size_t)4096 * 1024);
    unsigned short* w_ih1 = abf((size_t)4096 * 1024);
    unsigned short* w_hh1 = abf((size_t)4096 * 1024);
    unsigned short* w_e1  = abf((size_t)2048 * 1024);
    unsigned short* w_e2  = abf((size_t)4096 * 2048);
    unsigned short* w_e3  = abf((size_t)1024 * 4096);
    unsigned short* w_p1  = abf((size_t)2048 * 1024);
    unsigned short* w_p2  = abf((size_t)1024 * 2048);
    // bf16 activations
    unsigned short* z2_bf   = abf((size_t)BT * 512);
    unsigned short* mem_bf  = abf((size_t)100 * 512);
    unsigned short* z3_bf   = abf((size_t)BT * 1024);
    unsigned short* q_bf    = abf((size_t)BT * 512);
    unsigned short* keys_bf = abf((size_t)100 * 512);
    unsigned short* attn_bf = abf((size_t)BT * 128);
    unsigned short* valT_bf = abf((size_t)512 * 128);
    unsigned short* x0_bf   = abf((size_t)BT * 1536);
    unsigned short* h0seq   = abf((size_t)BT * 1024);
    unsigned short* h1seq   = abf((size_t)BT * 1024);
    unsigned short* hzero   = abf((size_t)16 * 1024);
    unsigned short* b1_bf   = abf((size_t)BT * 2048);
    unsigned short* b2_bf   = abf((size_t)BT * 4096);
    unsigned short* zf_bf   = abf((size_t)128 * 1024);
    unsigned short* t1_bf   = abf((size_t)128 * 2048);
    // f32 intermediates
    float* z3_f   = af32((size_t)BT * 1024);
    float* q_f    = af32((size_t)BT * 512);
    float* keys_f = af32((size_t)100 * 512);
    float* vals_f = af32((size_t)100 * 512);
    float* sc_f   = af32((size_t)BT * 128);
    float* read_f = af32((size_t)BT * 512);
    float* xg0_f  = af32((size_t)BT * 4096);
    float* xg1_f  = af32((size_t)BT * 4096);
    float* c0_f   = af32((size_t)16 * 1024);
    float* c1_f   = af32((size_t)16 * 1024);
    float* gb0_f  = af32(4096);
    float* gb1_f  = af32(4096);
    float* a1_f   = af32((size_t)BT * 2048);
    float* a2_f   = af32((size_t)BT * 4096);
    float* t1_f   = af32((size_t)16 * 2048);

    // ---- weight / input conversion to bf16
    launch_cvt(in_w,  w_in,  (size_t)1024 * 512, stream);
    launch_cvt(mq_w,  w_mq,  (size_t)512 * 1024, stream);
    launch_cvt(mk_w,  w_mk,  (size_t)512 * 512,  stream);
    launch_cvt(mv_w,  w_mv,  (size_t)512 * 512,  stream);
    launch_cvt(wih0,  w_ih0, (size_t)4096 * 1536, stream);
    launch_cvt(whh0,  w_hh0, (size_t)4096 * 1024, stream);
    launch_cvt(wih1,  w_ih1, (size_t)4096 * 1024, stream);
    launch_cvt(whh1,  w_hh1, (size_t)4096 * 1024, stream);
    launch_cvt(e1_w,  w_e1,  (size_t)2048 * 1024, stream);
    launch_cvt(e2_w,  w_e2,  (size_t)4096 * 2048, stream);
    launch_cvt(e3_w,  w_e3,  (size_t)1024 * 4096, stream);
    launch_cvt(p1_w,  w_p1,  (size_t)2048 * 1024, stream);
    launch_cvt(p2_w,  w_p2,  (size_t)1024 * 2048, stream);
    launch_cvt(z2,    z2_bf, (size_t)BT * 512,    stream);
    launch_cvt(memory, mem_bf, (size_t)100 * 512, stream);

    // zero state buffers (h0 init, c states)
    k_zero32<<<dim3((16 * 1024 * 2 / 4 + 255) / 256), dim3(256), 0, stream>>>((unsigned int*)hzero, (size_t)16 * 1024 * 2 / 4);
    k_zero32<<<dim3((16 * 1024 + 255) / 256), dim3(256), 0, stream>>>((unsigned int*)c0_f, (size_t)16 * 1024);
    k_zero32<<<dim3((16 * 1024 + 255) / 256), dim3(256), 0, stream>>>((unsigned int*)c1_f, (size_t)16 * 1024);

    // ---- z3 = z2 @ in_w^T + in_b
    launch_gemm(z2_bf, 512, w_in, 512, in_b, z3_f, 1024, BT, 1024, 512, 0, stream);
    launch_cvt(z3_f, z3_bf, (size_t)BT * 1024, stream);

    // ---- memory keys/values, query
    launch_gemm(mem_bf, 512, w_mk, 512, mk_b, keys_f, 512, 100, 512, 512, 0, stream);
    launch_gemm(mem_bf, 512, w_mv, 512, mv_b, vals_f, 512, 100, 512, 512, 0, stream);
    launch_gemm(z3_bf, 1024, w_mq, 1024, mq_b, q_f, 512, BT, 512, 1024, 0, stream);
    launch_cvt(q_f, q_bf, (size_t)BT * 512, stream);
    launch_cvt(keys_f, keys_bf, (size_t)100 * 512, stream);

    // ---- scores = q @ keys^T (N=100, ldc=128), softmax, read = attn @ values
    launch_gemm(q_bf, 512, keys_bf, 512, nullptr, sc_f, 128, BT, 100, 512, 0, stream);
    k_softmax<<<dim3(BT), dim3(128), 0, stream>>>(sc_f, attn_bf);
    k_valT<<<dim3((512 * 128 + 255) / 256), dim3(256), 0, stream>>>(vals_f, valT_bf);
    launch_gemm(attn_bf, 128, valT_bf, 128, nullptr, read_f, 512, BT, 512, 128, 0, stream);

    // ---- LSTM layer 0
    k_concat<<<dim3((unsigned)(((size_t)BT * 1536 + 255) / 256)), dim3(256), 0, stream>>>(z3_f, read_f, x0_bf, (size_t)BT * 1536);
    k_addvec<<<dim3(16), dim3(256), 0, stream>>>(bih0, bhh0, gb0_f, 4096);
    launch_gemm(x0_bf, 1536, w_ih0, 1536, gb0_f, xg0_f, 4096, BT, 4096, 1536, 0, stream);
    for (int t = 0; t < 512; ++t) {
        const unsigned short* hp = (t == 0) ? hzero : (h0seq + (size_t)(t - 1) * 1024);
        const int hps = (t == 0) ? 1024 : 512 * 1024;
        k_lstm_step<<<dim3(64), dim3(128), 0, stream>>>(hp, hps, w_hh0, xg0_f, c0_f, h0seq, t);
    }

    // ---- LSTM layer 1
    k_addvec<<<dim3(16), dim3(256), 0, stream>>>(bih1, bhh1, gb1_f, 4096);
    launch_gemm(h0seq, 1024, w_ih1, 1024, gb1_f, xg1_f, 4096, BT, 4096, 1024, 0, stream);
    for (int t = 0; t < 512; ++t) {
        const unsigned short* hp = (t == 0) ? hzero : (h1seq + (size_t)(t - 1) * 1024);
        const int hps = (t == 0) ? 1024 : 512 * 1024;
        k_lstm_step<<<dim3(64), dim3(128), 0, stream>>>(hp, hps, w_hh1, xg1_f, c1_f, h1seq, t);
    }

    // ---- JEPA encoder
    launch_gemm(h1seq, 1024, w_e1, 1024, e1_b, a1_f, 2048, BT, 2048, 1024, 0, stream);
    k_ln_gelu<<<dim3(BT), dim3(256), 0, stream>>>(a1_f, ln1_g, ln1_b, b1_bf, 2048);
    launch_gemm(b1_bf, 2048, w_e2, 2048, e2_b, a2_f, 4096, BT, 4096, 2048, 0, stream);
    k_ln_gelu<<<dim3(BT), dim3(256), 0, stream>>>(a2_f, ln2_g, ln2_b, b2_bf, 4096);
    launch_gemm(b2_bf, 4096, w_e3, 4096, e3_b, out_enc, 1024, BT, 1024, 4096, 0, stream);

    // ---- heads: mood (3) + pers (5)
    k_heads<<<dim3(BT), dim3(128), 0, stream>>>(out_enc, mood_w, mood_b, pers_w, pers_b, out_mood, out_pers);

    // ---- predictor: zf = enc[:, -1]; pred = gelu(zf@p1^T+b1) @ p2^T + b2
    k_zf<<<dim3((128 * 1024 + 255) / 256), dim3(256), 0, stream>>>(out_enc, zf_bf);
    launch_gemm(zf_bf, 1024, w_p1, 1024, p1_b, t1_f, 2048, 16, 2048, 1024, 1, stream);
    k_t1pad<<<dim3((128 * 2048 + 255) / 256), dim3(256), 0, stream>>>(t1_f, t1_bf);
    launch_gemm(t1_bf, 2048, w_p2, 2048, p2_b, out_pred, 1024, 16, 1024, 2048, 0, stream);
}
